// dsc_46437186404711
// MI455X (gfx1250) — matrix operand order repaired and checked by reading, not compiled, not run on hardware
//
#include <hip/hip_runtime.h>
#include <math.h>

// Problem constants (from reference)
#define B_   32
#define C_   128
#define H_   56
#define W_   56
#define HW_  (H_ * W_)      // 3136
#define K_   4
#define HID_ 33
#define TEMP_ 34.0f

#define LDSB_STRIDE 68      // padded row stride (dwords) -> conflict-free ds_load_b128

typedef __attribute__((ext_vector_type(16))) _Float16 v16h;
typedef __attribute__((ext_vector_type(8)))  float    v8f;

union AV { v16h v; unsigned int u[8]; };
union BV { v16h v; uint4 q[2]; };

__device__ __forceinline__ unsigned int pack_h2(float a, float b) {
    union { _Float16 h; unsigned short s; } ua, ub;
    ua.h = (_Float16)a; ub.h = (_Float16)b;
    return ((unsigned int)ub.s << 16) | (unsigned int)ua.s;
}

__device__ __forceinline__ unsigned short f2h_bits(float a) {
    union { _Float16 h; unsigned short s; } u;
    u.h = (_Float16)a;
    return u.s;
}

// ---------------------------------------------------------------------------
// Kernel 1: global average pool.  One block per (b,c) plane of 3136 floats.
// ---------------------------------------------------------------------------
__global__ __launch_bounds__(256) void pool_kernel(const float* __restrict__ x,
                                                   float* __restrict__ pooled) {
    const int bc = blockIdx.x;                 // b*C + c
    const float* p = x + (size_t)bc * HW_;
    float s = 0.0f;
    for (int i = threadIdx.x; i < HW_; i += 256) s += p[i];
    __shared__ float red[256];
    red[threadIdx.x] = s;
    __syncthreads();
    for (int st = 128; st > 0; st >>= 1) {
        if (threadIdx.x < st) red[threadIdx.x] += red[threadIdx.x + st];
        __syncthreads();
    }
    if (threadIdx.x == 0) pooled[bc] = red[0] * (1.0f / (float)HW_);
}

// ---------------------------------------------------------------------------
// Kernel 2: attention MLP + softmax + aggregated bias.  Single 128-thread block.
// ---------------------------------------------------------------------------
__global__ __launch_bounds__(128) void attn_kernel(const float* __restrict__ pooled,
                                                   const float* __restrict__ fc1_w,
                                                   const float* __restrict__ fc1_b,
                                                   const float* __restrict__ fc2_w,
                                                   const float* __restrict__ fc2_b,
                                                   const float* __restrict__ dw_bias,
                                                   float* __restrict__ attn,
                                                   float* __restrict__ aggb) {
    __shared__ float h_s[B_ * HID_];
    __shared__ float l_s[B_ * K_];
    __shared__ float a_s[B_ * K_];
    const int tid = threadIdx.x;

    // h = relu(pooled @ fc1^T + b1)
    for (int idx = tid; idx < B_ * HID_; idx += 128) {
        int b = idx / HID_, j = idx % HID_;
        float s = fc1_b[j];
        const float* pr = pooled + b * C_;
        const float* wr = fc1_w + j * C_;
        for (int c = 0; c < C_; ++c) s += pr[c] * wr[c];
        h_s[idx] = fmaxf(s, 0.0f);
    }
    __syncthreads();

    // logits = h @ fc2^T + b2
    if (tid < B_ * K_) {
        int b = tid >> 2, k = tid & 3;
        float s = fc2_b[k];
        const float* hr = h_s + b * HID_;
        const float* wr = fc2_w + k * HID_;
        for (int j = 0; j < HID_; ++j) s += hr[j] * wr[j];
        l_s[tid] = s;
    }
    __syncthreads();

    // softmax(logits / TEMP) per batch
    if (tid < B_) {
        int b = tid;
        float m = l_s[b * 4];
        for (int k = 1; k < 4; ++k) m = fmaxf(m, l_s[b * 4 + k]);
        float e[4], sum = 0.0f;
        for (int k = 0; k < 4; ++k) { e[k] = __expf((l_s[b * 4 + k] - m) * (1.0f / TEMP_)); sum += e[k]; }
        float inv = 1.0f / sum;
        for (int k = 0; k < 4; ++k) {
            float a = e[k] * inv;
            a_s[b * 4 + k] = a;
            attn[b * 4 + k] = a;
        }
    }
    __syncthreads();

    // agg_b = attn @ dw_bias
    for (int idx = tid; idx < B_ * C_; idx += 128) {
        int b = idx >> 7, o = idx & (C_ - 1);
        float s = 0.0f;
        for (int k = 0; k < 4; ++k) s += a_s[b * 4 + k] * dw_bias[k * C_ + o];
        aggb[idx] = s;
    }
}

// ---------------------------------------------------------------------------
// Kernel 3: per-sample mixed 1x1 weight, stored f16 row-major [B][o][i].
// ---------------------------------------------------------------------------
__global__ __launch_bounds__(256) void aggw_kernel(const float* __restrict__ attn,
                                                   const float* __restrict__ dw_weight,
                                                   unsigned short* __restrict__ aggw) {
    const int e = blockIdx.x * 256 + threadIdx.x;     // e < B*C*C = 524288
    const int b = e >> 14;
    const int oi = e & 16383;
    float s = 0.0f;
#pragma unroll
    for (int k = 0; k < 4; ++k) s += attn[b * 4 + k] * dw_weight[k * 16384 + oi];
    aggw[e] = f2h_bits(s);
}

// ---------------------------------------------------------------------------
// Kernel 4: per-sample channel-mix GEMM via WMMA.
//   y[b] (128 x 3136) = aggw[b] (128x128, f16) * x[b] (128x3136, ->f16) + aggb[b]
// Block: 256 threads (8 waves); block tile = 128(M) x 64(N), K = 128.
// LDS: x-tile packed as half2 per K-pair, TRANSPOSED layout ldsB[n][kpair]
// with padded stride 68 so each lane's 8 B-operand dwords are two contiguous 4-dword runs 8 dwords apart
// (2x ds_load_b128, conflict-free: lane j covers banks j*4..j*4+3 mod 64).
// ---------------------------------------------------------------------------
__global__ __launch_bounds__(256) void gemm_kernel(const float* __restrict__ x,
                                                   const unsigned int* __restrict__ aggw_pk, // dword view of f16 [B][128][128]
                                                   const float* __restrict__ aggb,
                                                   float* __restrict__ y) {
    __shared__ unsigned int ldsB[64 * LDSB_STRIDE];   // ~17 KB
    const int b  = blockIdx.y;
    const int n0 = blockIdx.x * 64;
    const int tid = threadIdx.x;
    const float* xb = x + (size_t)b * C_ * HW_;

    // Prefetch next N-tile of x into L2 (emits global_prefetch_b8)
    if (blockIdx.x + 1 < gridDim.x)
        __builtin_prefetch(xb + (tid >> 2) * HW_ + n0 + 64 + (tid & 3) * 16, 0, 1);

    // Stage x tile into LDS: pack (k=2p, 2p+1) as half2, store at [n][p].
    for (int e = tid; e < 64 * 64; e += 256) {
        int p = e >> 6, n = e & 63;
        const float* s = xb + (p * 2) * HW_ + n0 + n;
        ldsB[n * LDSB_STRIDE + p] = pack_h2(s[0], s[HW_]);
    }
    __syncthreads();

    const int wave = tid >> 5;
    const int lane = tid & 31;
    const int lhi  = lane >> 4;          // 0 for lanes 0-15, 1 for 16-31
    const int mlo  = lane & 15;

    // A operand: 16(M) x 128(K) slice of aggw, ISA 16-bit A layout.
    // VGPR v, half h -> K = kc*32 + ((v&4)<<2) + (lhi<<3) + ((v&3)<<1) + h
    AV A[4];
    const unsigned int* awb = aggw_pk + ((size_t)b * C_ + wave * 16 + mlo) * (C_ / 2);
#pragma unroll
    for (int kc = 0; kc < 4; ++kc) {
#pragma unroll
        for (int v = 0; v < 8; ++v) {
            int kb = kc * 32 + ((v & 4) << 2) + (lhi << 3) + ((v & 3) << 1);
            A[kc].u[v] = awb[kb >> 1];
        }
    }

    // Per-lane bias values for the 8 C/D rows this lane owns.
    float bias[8];
#pragma unroll
    for (int r = 0; r < 8; ++r) bias[r] = aggb[b * C_ + wave * 16 + lhi * 8 + r];

#pragma unroll
    for (int nt = 0; nt < 4; ++nt) {
        v8f acc = {0.f, 0.f, 0.f, 0.f, 0.f, 0.f, 0.f, 0.f};
        const int col = nt * 16 + mlo;
#pragma unroll
        for (int kc = 0; kc < 4; ++kc) {
            // B operand: 32(K) x 16(N); lane needs kpairs [kc*16+lhi*4 .. +3] and
            // [kc*16+8+lhi*4 .. +3] at its col -> two ds_load_b128.
            BV Bv;
            const uint4* q = reinterpret_cast<const uint4*>(
                &ldsB[col * LDSB_STRIDE + kc * 16 + (lhi << 2)]);
            Bv.q[0] = q[0];
            Bv.q[1] = q[2];
            acc = __builtin_amdgcn_wmma_f32_16x16x32_f16(
                false, A[kc].v, false, Bv.v, (short)0, acc, false, false);
        }
        // C/D layout: VGPR r -> M = wave*16 + lhi*8 + r, N = mlo
        float* yo = y + (size_t)b * C_ * HW_ + (size_t)(wave * 16 + lhi * 8) * HW_ + n0 + col;
#pragma unroll
        for (int r = 0; r < 8; ++r) yo[(size_t)r * HW_] = acc[r] + bias[r];
    }
}

// ---------------------------------------------------------------------------
// Kernel 5: depthwise 3x3 at dilations {1,3,5}, SiLU, summed.
// Block handles a 56x4 tile of one (b,c) plane; halo = 5 each side in LDS.
// ---------------------------------------------------------------------------
__global__ __launch_bounds__(256) void dwconv_kernel(const float* __restrict__ y,
                                                     const float* __restrict__ ds_w,
                                                     const float* __restrict__ ds_b,
                                                     float* __restrict__ out) {
    __shared__ float tile[14 * 66];                   // (4+10) rows x (56+10) cols
    const int ty = blockIdx.x;                        // 0..13 -> rows 4*ty..4*ty+3
    const int c  = blockIdx.y;
    const int b  = blockIdx.z;
    const int tid = threadIdx.x;
    const float* yp = y + ((size_t)b * C_ + c) * HW_;

    for (int e = tid; e < 14 * 66; e += 256) {
        int r = e / 66, col = e % 66;
        int gy = ty * 4 + r - 5;
        int gx = col - 5;
        float v = 0.0f;
        if (gy >= 0 && gy < H_ && gx >= 0 && gx < W_) v = yp[gy * W_ + gx];
        tile[e] = v;
    }
    __syncthreads();

    if (tid < 56 * 4) {
        const int r  = tid / 56;                      // 0..3
        const int xc = tid % 56;
        float w[9];
#pragma unroll
        for (int i = 0; i < 9; ++i) w[i] = ds_w[c * 9 + i];
        const float bias = ds_b[c];
        const int base = (r + 5) * 66 + (xc + 5);

        float acc = 0.0f;
        const int dils[3] = {1, 3, 5};
#pragma unroll
        for (int d = 0; d < 3; ++d) {
            const int dil = dils[d];
            float s = bias;
#pragma unroll
            for (int ky = 0; ky < 3; ++ky)
#pragma unroll
                for (int kx = 0; kx < 3; ++kx)
                    s += w[ky * 3 + kx] * tile[base + (ky - 1) * dil * 66 + (kx - 1) * dil];
            acc += s * (1.0f / (1.0f + __expf(-s)));  // SiLU
        }
        const int gy = ty * 4 + r;
        out[((size_t)b * C_ + c) * HW_ + gy * W_ + xc] = acc;
    }
}

// ---------------------------------------------------------------------------
// Host launcher
// ---------------------------------------------------------------------------
extern "C" void kernel_launch(void* const* d_in, const int* in_sizes, int n_in,
                              void* d_out, int out_size, void* d_ws, size_t ws_size,
                              hipStream_t stream) {
    const float* x      = (const float*)d_in[0];
    const float* dw_w   = (const float*)d_in[1];
    const float* dw_b   = (const float*)d_in[2];
    const float* fc1_w  = (const float*)d_in[3];
    const float* fc1_b  = (const float*)d_in[4];
    const float* fc2_w  = (const float*)d_in[5];
    const float* fc2_b  = (const float*)d_in[6];
    const float* ds_w   = (const float*)d_in[7];
    const float* ds_b   = (const float*)d_in[8];
    float* out = (float*)d_out;

    // Workspace layout (float offsets):
    //   pooled : [0, 4096)
    //   attn   : [4096, 4224)
    //   aggb   : [4224, 8320)
    //   aggw   : f16 x 524288 at float offset 8320  (1 MB)
    //   y      : float x 12,845,056 after aggw
    float* ws_f = (float*)d_ws;
    float* pooled = ws_f;
    float* attn   = ws_f + 4096;
    float* aggb   = ws_f + 4224;
    unsigned short* aggw = (unsigned short*)(ws_f + 8320);
    float* y = (float*)(aggw + (size_t)B_ * C_ * C_);

    pool_kernel<<<B_ * C_, 256, 0, stream>>>(x, pooled);
    attn_kernel<<<1, 128, 0, stream>>>(pooled, fc1_w, fc1_b, fc2_w, fc2_b, dw_b, attn, aggb);
    aggw_kernel<<<(B_ * C_ * C_) / 256, 256, 0, stream>>>(attn, dw_w, aggw);
    gemm_kernel<<<dim3(HW_ / 64, B_), 256, 0, stream>>>(x, (const unsigned int*)aggw, aggb, y);
    dwconv_kernel<<<dim3(H_ / 4, C_, B_), 256, 0, stream>>>(y, ds_w, ds_b, out);
}
